// DocumentRetriever_18399639896794
// MI455X (gfx1250) — matrix operand order repaired and checked by reading, not compiled, not run on hardware
//
#include <hip/hip_runtime.h>
#include <hip/hip_bf16.h>
#include <math.h>
#include <stdint.h>

typedef __attribute__((ext_vector_type(16))) _Float16 v16h;
typedef __attribute__((ext_vector_type(8)))  _Float16 v8h;
typedef __attribute__((ext_vector_type(8)))  float    v8f;
typedef __attribute__((ext_vector_type(4)))  float    v4f;
typedef __attribute__((ext_vector_type(4)))  unsigned v4u;
typedef __attribute__((ext_vector_type(8)))  int      v8i;
typedef __attribute__((ext_vector_type(4)))  int      v4i;

// ---------------------------------------------------------------------------
// Vectorized f32 -> f16 conversion, 8 elements per thread (n must be %8 == 0)
// ---------------------------------------------------------------------------
__global__ __launch_bounds__(256) void cvt_f32_to_f16_v8(const float* __restrict__ in,
                                                         _Float16* __restrict__ out,
                                                         long n) {
  long i = ((long)blockIdx.x * 256 + threadIdx.x) * 8;
  if (i < n) {
    v8f v;
    *((v4f*)&v)     = *(const v4f*)(in + i);
    *((v4f*)&v + 1) = *(const v4f*)(in + i + 4);
    *(v8h*)(out + i) = __builtin_convertvector(v, v8h);
  }
}

// ---------------------------------------------------------------------------
// LDS-tiled transpose + convert:  out[c*rows + r] = (f16) in[r*cols + c]
// blockDim (32,8); grid (cols/32, rows/32, batch); batch stride = rows*cols.
// ---------------------------------------------------------------------------
__global__ __launch_bounds__(256) void transpose_cvt_kernel(const float* __restrict__ in,
                                                            _Float16* __restrict__ out,
                                                            int rows, int cols) {
  __shared__ float tile[32][33];
  long zoff = (long)blockIdx.z * rows * cols;
  in  += zoff;
  out += zoff;
  int bx = blockIdx.x * 32;   // column base (input)
  int by = blockIdx.y * 32;   // row base (input)
  int tx = threadIdx.x, ty = threadIdx.y;
#pragma unroll
  for (int j = 0; j < 32; j += 8)
    tile[ty + j][tx] = in[(long)(by + ty + j) * cols + (bx + tx)];
  __syncthreads();
#pragma unroll
  for (int j = 0; j < 32; j += 8)
    out[(long)(bx + ty + j) * rows + (by + tx)] = (_Float16)tile[tx][ty + j];
}

// ---------------------------------------------------------------------------
// TDM: issue a 2D TENSOR_LOAD_TO_LDS of a 16-row x 32-col f16 tile.
// Source rows are K-contiguous with row stride K elements; dest is a packed
// [16][32] f16 LDS slice. Descriptor per CDNA5 ISA §8.3/8.4:
//   g0: count=1 | lds_addr | global_addr[56:0] | type=2
//   g1: data_size=1 (2B) | tensor_dim0=K | tensor_dim1=16 | tile=32x16
//       | tensor_dim0_stride=K ; groups 2/3 zero (2D tensor).
// Tracked by TENSORcnt (s_wait_tensorcnt).
// ---------------------------------------------------------------------------
__device__ __forceinline__ void tdm_load_tile16x32(unsigned lds_off,
                                                   const _Float16* gptr, int K) {
  unsigned long long ga = (unsigned long long)(uintptr_t)gptr;
  v4u g0 = { 1u,                                   // count=1, user mode
             lds_off,                              // lds_addr
             (unsigned)ga,                         // global_addr[31:0]
             (unsigned)((ga >> 32) & 0x1FFFFFFu) | (2u << 30) };  // [56:32] | type=2
  unsigned uk = (unsigned)K;
  v8i g1 = { (int)(1u << 16),                      // data_size = 2 bytes
             (int)((uk & 0xFFFFu) << 16),          // tensor_dim0[15:0]
             (int)((uk >> 16) | (16u << 16)),      // tensor_dim0[31:16] | tensor_dim1=16
             (int)(32u << 16),                     // tile_dim0 = 32
             16,                                   // tile_dim1 = 16, tile_dim2 = 0
             (int)uk,                              // tensor_dim0_stride[31:0] = K
             0, 0 };
  v4i z4 = {0, 0, 0, 0};
#if __clang_major__ >= 23
  v8i z8 = {0, 0, 0, 0, 0, 0, 0, 0};
  __builtin_amdgcn_tensor_load_to_lds(g0, g1, z4, z4, z8, 0);
#else
  __builtin_amdgcn_tensor_load_to_lds(g0, g1, z4, z4, 0);
#endif
}

// ---------------------------------------------------------------------------
// WMMA GEMM with TDM-staged, double-buffered LDS B tiles:
//   C[M,N] = scale * (A[M,K] * Bt[N,K]^T) + bias
//   A  : f16 row-major [M,K], batch stride sA (loaded direct from global)
//   Bt : f16 row-major [N,K] (B^T, contiguous K), batch stride sB (via TDM->LDS)
//   Cf : optional f32 out [M,N], batch stride sC;  Ch : optional f16 out
// Block = 128 threads = 4 waves -> 128x64 output. Each wave computes a 32x64
// block (MT=2 x NT=4 accumulators); the 64x32 B k-tile is shared through LDS,
// each wave DMAs its own 16-row slice per k-step (double buffered).
// Fragment layouts per CDNA5 ISA 7.12.2:
//   A: lane m=L&15, half=L>>4; halves 0..7  <- K = k0 + half*8 + (0..7)
//                              halves 8..15 <- K = k0 + 16 + half*8 + (0..7)
//   B: lane n=L&15; same K mapping as A (halves 0..7 <- k0 + half*8, 8..15 <- k0 + 16 + half*8)
//   C: VGPR r -> (m = r + (L>>4)*8, n = L&15)
// Requires M % 128 == 0, N % 64 == 0, K % 32 == 0.
// ---------------------------------------------------------------------------
#define MT 2
#define NT 4
__global__ __launch_bounds__(128) void gemm_wmma_f16_tdm(
    const _Float16* __restrict__ A, const _Float16* __restrict__ Bt,
    float* __restrict__ Cf, _Float16* __restrict__ Ch,
    const float* __restrict__ bias,
    int M, int N, int K, long sA, long sB, long sC, float scale) {
  __shared__ _Float16 ldsB[2][64][32];   // double-buffered 64x32 f16 B tile

  int lane   = threadIdx.x & 31;
  int waveId = __builtin_amdgcn_readfirstlane((int)(threadIdx.x >> 5));
  int tm = blockIdx.x * 4 + waveId;      // this wave's 32-row block
  int tn = blockIdx.y;                   // shared 64-col block
  int b  = blockIdx.z;
  int half = lane >> 4;
  int lm   = lane & 15;

  const _Float16* aBase = A  + (long)b * sA + (long)(tm * 32 + lm) * K;
  const _Float16* bTile = Bt + (long)b * sB + (long)(tn * 64 + waveId * 16) * K;
  const long tStep = 16L * K;
  int aOff = half * 8;
  int bOff = half * 8;

  unsigned ldsSlice[2];
  ldsSlice[0] = (unsigned)(uintptr_t)&ldsB[0][waveId * 16][0];
  ldsSlice[1] = (unsigned)(uintptr_t)&ldsB[1][waveId * 16][0];

  // Prologue: DMA the first B k-tile slice.
  tdm_load_tile16x32(ldsSlice[0], bTile, K);

  v8f acc[MT][NT] = {};
  for (int k0 = 0; k0 < K; k0 += 32) {
    int cur = (k0 >> 5) & 1;
    __builtin_amdgcn_s_wait_tensorcnt(0);  // own slice of buf[cur] landed
    __syncthreads();                       // all slices landed; buf[cur^1] free
    if (k0 + 32 < K)
      tdm_load_tile16x32(ldsSlice[cur ^ 1], bTile + (k0 + 32), K);

    v16h a[MT], bf[NT];
#pragma unroll
    for (int i = 0; i < MT; ++i) {
      const _Float16* p = aBase + i * tStep + k0 + aOff;
      *((v8h*)&a[i])     = *(const v8h*)(p);
      *((v8h*)&a[i] + 1) = *(const v8h*)(p + 16);
    }
    __builtin_prefetch(aBase + k0 + 32, 0, 1);          // next A k-block
    __builtin_prefetch(aBase + tStep + k0 + 32, 0, 1);
#pragma unroll
    for (int j = 0; j < NT; ++j) {
      const _Float16* p = &ldsB[cur][j * 16 + lm][bOff];
      *((v8h*)&bf[j])     = *(const v8h*)(p);      // ds_load_b128
      *((v8h*)&bf[j] + 1) = *(const v8h*)(p + 16);
    }
#pragma unroll
    for (int i = 0; i < MT; ++i)
#pragma unroll
      for (int j = 0; j < NT; ++j)
        acc[i][j] = __builtin_amdgcn_wmma_f32_16x16x32_f16(
            false, a[i], false, bf[j], (short)0, acc[i][j], false, false);
  }

#pragma unroll
  for (int j = 0; j < NT; ++j) {
    int col = tn * 64 + j * 16 + lm;
    float bv = bias ? bias[col] : 0.0f;
#pragma unroll
    for (int i = 0; i < MT; ++i) {
#pragma unroll
      for (int r = 0; r < 8; ++r) {
        int row = tm * 32 + i * 16 + r + half * 8;
        float v = acc[i][j][r] * scale + bv;
        long off = (long)b * sC + (long)row * N + col;
        if (Cf) Cf[off] = v;
        if (Ch) Ch[off] = (_Float16)v;
      }
    }
  }
}

// ---------------------------------------------------------------------------
// Masked softmax over D, in place on f32 scores; also emits f16 copy.
// grid (Q, B), 256 threads per row.
// ---------------------------------------------------------------------------
__global__ __launch_bounds__(256) void softmax_mask(
    float* __restrict__ scores,     // [B*Q, D] in/out
    _Float16* __restrict__ attn16,  // [B*Q, D] out
    const int* __restrict__ mask,   // [B, D]
    int Q, int D) {
  int q = blockIdx.x, b = blockIdx.y;
  long rowOff = ((long)b * Q + q) * D;
  float* s = scores + rowOff;
  _Float16* o = attn16 + rowOff;
  const int* mk = mask + (long)b * D;
  __shared__ float red[256];
  int tid = threadIdx.x;

  float mx = -INFINITY;
  for (int i = tid; i < D; i += 256)
    if (mk[i]) mx = fmaxf(mx, s[i]);
  red[tid] = mx;
  __syncthreads();
  for (int off = 128; off > 0; off >>= 1) {
    if (tid < off) red[tid] = fmaxf(red[tid], red[tid + off]);
    __syncthreads();
  }
  mx = red[0];
  __syncthreads();

  float sum = 0.0f;
  for (int i = tid; i < D; i += 256)
    if (mk[i]) sum += expf(s[i] - mx);
  red[tid] = sum;
  __syncthreads();
  for (int off = 128; off > 0; off >>= 1) {
    if (tid < off) red[tid] += red[tid + off];
    __syncthreads();
  }
  float inv = 1.0f / red[0];

  for (int i = tid; i < D; i += 256) {
    float p = mk[i] ? expf(s[i] - mx) * inv : 0.0f;
    s[i] = p;
    o[i] = (_Float16)p;
  }
}

// ---------------------------------------------------------------------------
// Launch
// ---------------------------------------------------------------------------
extern "C" void kernel_launch(void* const* d_in, const int* in_sizes, int n_in,
                              void* d_out, int out_size, void* d_ws, size_t ws_size,
                              hipStream_t stream) {
  const float* query = (const float*)d_in[0];  // [B,Q,H]
  const float* docs  = (const float*)d_in[1];  // [B,D,H]
  const int*   mask  = (const int*)d_in[2];    // [B,D]
  const float* projW = (const float*)d_in[3];  // [H,H]
  const float* projB = (const float*)d_in[4];  // [H]

  const int B = 8, Q = 512, D = 4096, H = 1024;
  const float scale = 0.03125f;  // 1/sqrt(1024)

  float* retrieved = (float*)d_out;                       // [B,Q,H]
  float* attnOut   = (float*)d_out + (long)B * Q * H;     // [B,Q,D]

  // Workspace layout (~146 MiB); attn16 aliases doc16 (dead after scores GEMM).
  char* ws = (char*)d_ws;
  _Float16* query16 = (_Float16*)ws; ws += (long)B * Q * H * 2;
  _Float16* q16     = (_Float16*)ws; ws += (long)B * Q * H * 2;
  _Float16* wt16    = (_Float16*)ws; ws += (long)H * H * 2;
  _Float16* doc16   = (_Float16*)ws; ws += (long)B * D * H * 2;
  _Float16* docT16  = (_Float16*)ws; ws += (long)B * D * H * 2;
  _Float16* attn16  = doc16;

  long nq = (long)B * Q * H;
  long nd = (long)B * D * H;

  // 1) f16 copies of query and docs (8 elems/thread, b128 traffic)
  cvt_f32_to_f16_v8<<<dim3((unsigned)((nq / 8 + 255) / 256)), 256, 0, stream>>>(query, query16, nq);
  cvt_f32_to_f16_v8<<<dim3((unsigned)((nd / 8 + 255) / 256)), 256, 0, stream>>>(docs, doc16, nd);

  // 2) W^T f16 (so GEMM B-operand reads contiguous K)
  transpose_cvt_kernel<<<dim3(H / 32, H / 32, 1), dim3(32, 8), 0, stream>>>(projW, wt16, H, H);

  // 3) docs^T f16 per batch: [D,H] -> [H,D]
  transpose_cvt_kernel<<<dim3(H / 32, D / 32, B), dim3(32, 8), 0, stream>>>(docs, docT16, D, H);

  // 4) Projection: q16 = query16 @ W + b   (M = B*Q, batch = 1)
  gemm_wmma_f16_tdm<<<dim3((B * Q) / 128, H / 64, 1), 128, 0, stream>>>(
      query16, wt16, nullptr, q16, projB,
      B * Q, H, H, 0L, 0L, 0L, 1.0f);

  // 5) Scores into attention region of d_out: scale * q @ docs^T
  gemm_wmma_f16_tdm<<<dim3(Q / 128, D / 64, B), 128, 0, stream>>>(
      q16, doc16, attnOut, nullptr, nullptr,
      Q, D, H, (long)Q * H, (long)D * H, (long)Q * D, scale);

  // 6) Masked softmax in place (f32) + f16 copy for the AV GEMM
  softmax_mask<<<dim3(Q, B), 256, 0, stream>>>(attnOut, attn16, mask, Q, D);

  // 7) retrieved = attn @ docs
  gemm_wmma_f16_tdm<<<dim3(Q / 128, H / 64, B), 128, 0, stream>>>(
      attn16, docT16, retrieved, nullptr, nullptr,
      Q, H, D, (long)Q * D, (long)D * H, (long)Q * H, 1.0f);
}
